// MultiHeadAttention_4595615007055
// MI455X (gfx1250) — matrix operand order repaired and checked by reading, not compiled, not run on hardware
//
#include <hip/hip_runtime.h>
#include <hip/hip_bf16.h>
#include <stdint.h>

typedef __bf16 v16bf __attribute__((ext_vector_type(16)));
typedef __bf16 v8bf  __attribute__((ext_vector_type(8)));
typedef float  v8f   __attribute__((ext_vector_type(8)));

#define DMODEL 1024
#define SEQ    2048
#define NBATCH 4
#define NHEAD  16
#define DK     64

__device__ __forceinline__ v16bf cat16(v8bf lo, v8bf hi) {
  return __builtin_shufflevector(lo, hi, 0,1,2,3,4,5,6,7,8,9,10,11,12,13,14,15);
}
__device__ __forceinline__ v8f wmma_bf16(v16bf a, v16bf b, v8f c) {
  return __builtin_amdgcn_wmma_f32_16x16x32_bf16(false, a, false, b, (short)0, c, false, false);
}
__device__ __forceinline__ v8f v8f_zero() {
  v8f z = {0.f,0.f,0.f,0.f,0.f,0.f,0.f,0.f};
  return z;
}
__device__ __forceinline__ v8bf cvt8(float4 a, float4 b) {
  v8bf h = { (__bf16)a.x,(__bf16)a.y,(__bf16)a.z,(__bf16)a.w,
             (__bf16)b.x,(__bf16)b.y,(__bf16)b.z,(__bf16)b.w };
  return h;
}

// CDNA5 async DMA: copy 16 bytes global -> LDS per lane, tracked by ASYNCcnt.
// vdst VGPR carries the LDS byte offset (generic shared ptr low 32 bits),
// vaddr is the 64-bit global address.
__device__ __forceinline__ void async_cp16(__bf16* ldst, const __bf16* gsrc) {
  asm volatile("global_load_async_to_lds_b128 %0, %1, off"
               :: "v"((unsigned)(uintptr_t)ldst), "v"(gsrc)
               : "memory");
}

// ---------------------------------------------------------------------------
// Elementwise f32 -> bf16 (n must be a multiple of 8; all our sizes are).
// ---------------------------------------------------------------------------
__global__ __launch_bounds__(256) void cvt_f32_bf16(const float* __restrict__ in,
                                                    __bf16* __restrict__ out, int n) {
  const int i = (blockIdx.x * 256 + threadIdx.x) * 8;
  if (i < n) {
    float4 a = *(const float4*)(in + i);
    float4 b = *(const float4*)(in + i + 4);
    *(v8bf*)(out + i) = cvt8(a, b);
  }
}

// ---------------------------------------------------------------------------
// All-bf16 projection GEMM:  Out[m,n] = sum_k A[m,k] * W[n,k] + bias[n]
// M = 8192, N = K = 1024. Block tile 128x128, 8 waves of 2x4 16x16 tiles.
// Tiles staged with global_load_async_to_lds_b128 (ASYNCcnt), double-buffered:
// 4 async instructions per wave per 128x32 tile-pair, one tile in flight.
// OUT_MODE: 0 = f32 [M, DMODEL]              (final output projection)
//           1 = bf16 [B, H, S, DK]           (Q, K for attention)
//           2 = bf16 [B, H, DK, S] transposed (V^T for attention B-frags)
// ---------------------------------------------------------------------------
template<int OUT_MODE>
__global__ __launch_bounds__(256) void proj_gemm(const __bf16* __restrict__ A,
                                                 const __bf16* __restrict__ W,
                                                 const float* __restrict__ bias,
                                                 void* __restrict__ Outp)
{
  constexpr int LDT = 40;                    // padded bf16 row stride (20 banks)
  __shared__ __bf16 As[2][128 * LDT];
  __shared__ __bf16 Bs[2][128 * LDT];

  const int t     = threadIdx.x;
  const int w     = t >> 5;
  const int lane  = t & 31;
  const int lhalf = lane >> 4;
  const int lmod  = lane & 15;
  const int wm    = (w & 3) * 32;            // wave rows within block tile
  const int wn    = (w >> 2) * 64;           // wave cols within block tile
  const int m0    = blockIdx.x * 128;
  const int n0    = blockIdx.y * 128;
  const int lrow  = t >> 1;                  // cooperative-copy row (0..127)
  const int lkh   = (t & 1) * 16;            // k half (0 or 16)

  const __bf16* agp = A + (size_t)(m0 + lrow) * DMODEL + lkh;
  const __bf16* wgp = W + (size_t)(n0 + lrow) * DMODEL + lkh;
  const int lds_off = lrow * LDT + lkh;

  // prologue: stage k-tile 0 into buffer 0
  async_cp16(&As[0][lds_off],     agp);
  async_cp16(&As[0][lds_off + 8], agp + 8);
  async_cp16(&Bs[0][lds_off],     wgp);
  async_cp16(&Bs[0][lds_off + 8], wgp + 8);

  v8f acc[2][4];
  #pragma unroll
  for (int i = 0; i < 2; i++)
    #pragma unroll
    for (int j = 0; j < 4; j++) acc[i][j] = v8f_zero();

  int cur = 0;
  for (int k0 = 0; k0 < DMODEL; k0 += 32) {
    if (k0 + 32 < DMODEL) {
      // stage next tile into the other buffer while we compute this one
      const int nb = cur ^ 1;
      async_cp16(&As[nb][lds_off],     agp + k0 + 32);
      async_cp16(&As[nb][lds_off + 8], agp + k0 + 40);
      async_cp16(&Bs[nb][lds_off],     wgp + k0 + 32);
      async_cp16(&Bs[nb][lds_off + 8], wgp + k0 + 40);
      asm volatile("s_wait_asynccnt 4" ::: "memory");   // current tile landed
    } else {
      asm volatile("s_wait_asynccnt 0" ::: "memory");
    }
    __syncthreads();

    // ---- A fragments: lane holds row lane%16, K runs {h*8..+7, +16} ----
    v16bf af[2];
    #pragma unroll
    for (int i = 0; i < 2; i++) {
      const __bf16* p = &As[cur][(wm + i * 16 + lmod) * LDT + lhalf * 8];
      af[i] = cat16(*(const v8bf*)p, *(const v8bf*)(p + 16));
    }
    // ---- B fragments: lane holds col lane%16, K runs {h*8..+7, +16} ----
    #pragma unroll
    for (int j = 0; j < 4; j++) {
      const __bf16* p = &Bs[cur][(wn + j * 16 + lmod) * LDT + lhalf * 8];
      v16bf bf = cat16(*(const v8bf*)p, *(const v8bf*)(p + 16));
      #pragma unroll
      for (int i = 0; i < 2; i++) acc[i][j] = wmma_bf16(af[i], bf, acc[i][j]);
    }
    __syncthreads();   // all waves done reading before next async overwrite
    cur ^= 1;
  }

  // ---- epilogue: C/D layout row = r + 8*lhalf, col = lane%16 ----
  #pragma unroll
  for (int i = 0; i < 2; i++) {
    #pragma unroll
    for (int j = 0; j < 4; j++) {
      const int col = n0 + wn + j * 16 + lmod;
      const float bv = bias[col];
      #pragma unroll
      for (int r = 0; r < 8; r++) {
        const int row = m0 + wm + i * 16 + lhalf * 8 + r;
        const float v = acc[i][j][r] + bv;
        if (OUT_MODE == 0) {
          ((float*)Outp)[(size_t)row * DMODEL + col] = v;
        } else {
          const int b = row >> 11, s = row & (SEQ - 1);
          const int h = col >> 6,  d = col & (DK - 1);
          if (OUT_MODE == 1)
            ((__bf16*)Outp)[(((size_t)(b * NHEAD + h)) * SEQ + s) * DK + d] = (__bf16)v;
          else
            ((__bf16*)Outp)[(((size_t)(b * NHEAD + h)) * DK + d) * SEQ + s] = (__bf16)v;
        }
      }
    }
  }
}

// ---------------------------------------------------------------------------
// Flash attention (non-causal). One wave = 16 query rows of one (b,h).
// K [B,H,S,DK] and V^T [B,H,DK,S] fragments loaded straight from global
// (entire K+V = 32 MB bf16, resident in the 192 MB L2). P goes through LDS
// to convert C/D register layout -> A-fragment layout.
// ---------------------------------------------------------------------------
__global__ __launch_bounds__(256) void attn_kernel(const __bf16* __restrict__ q,
                                                   const __bf16* __restrict__ k,
                                                   const __bf16* __restrict__ vt,
                                                   __bf16* __restrict__ o)
{
  constexpr int PST = 136;                  // padded bf16 stride for P tile
  __shared__ __bf16 P[8 * 16 * PST];
  const int t     = threadIdx.x;
  const int w     = t >> 5;
  const int lane  = t & 31;
  const int lhalf = lane >> 4;
  const int lmod  = lane & 15;
  const int bh    = blockIdx.y;                    // 0..63
  const int qbase = blockIdx.x * 128 + w * 16;     // wave's first query row
  const __bf16* qp = q  + ((size_t)bh * SEQ + qbase) * DK;
  const __bf16* kp = k  + (size_t)bh * SEQ * DK;
  const __bf16* vp = vt + (size_t)bh * DK * SEQ;
  __bf16* pw = &P[w * 16 * PST];

  // Q A-fragments over dk = 64 (2 k-steps), kept in registers all kernel
  v16bf qf[2];
  #pragma unroll
  for (int ks = 0; ks < 2; ks++) {
    const __bf16* p = qp + (size_t)lmod * DK + ks * 32 + lhalf * 8;
    qf[ks] = cat16(*(const v8bf*)p, *(const v8bf*)(p + 16));
  }

  float m_i[8], l_i[8];
  #pragma unroll
  for (int r = 0; r < 8; r++) { m_i[r] = -3.0e38f; l_i[r] = 0.f; }
  v8f oacc[4];
  #pragma unroll
  for (int nv = 0; nv < 4; nv++) oacc[nv] = v8f_zero();

  for (int kt = 0; kt < SEQ; kt += 128) {
    // ---- scores: 8 tiles of 16x16, K B-frags direct from global/L2 ----
    v8f sc[8];
    #pragma unroll
    for (int nt = 0; nt < 8; nt++) {
      v8f a = v8f_zero();
      #pragma unroll
      for (int ks = 0; ks < 2; ks++) {
        const __bf16* p = kp + ((size_t)(kt + nt * 16 + lmod)) * DK + ks * 32 + lhalf * 8;
        v16bf bf = cat16(*(const v8bf*)p, *(const v8bf*)(p + 16));
        a = wmma_bf16(qf[ks], bf, a);
      }
      sc[nt] = a * 0.125f;                    // 1/sqrt(DK)
    }

    // ---- online softmax: row stats within 16-lane halves ----
    float mnew[8], alpha[8];
    #pragma unroll
    for (int r = 0; r < 8; r++) {
      float mx = sc[0][r];
      #pragma unroll
      for (int nt = 1; nt < 8; nt++) mx = fmaxf(mx, sc[nt][r]);
      mx = fmaxf(mx, __shfl_xor(mx, 1));
      mx = fmaxf(mx, __shfl_xor(mx, 2));
      mx = fmaxf(mx, __shfl_xor(mx, 4));
      mx = fmaxf(mx, __shfl_xor(mx, 8));
      mnew[r]  = fmaxf(m_i[r], mx);
      alpha[r] = exp2f((m_i[r] - mnew[r]) * 1.44269504f);
      m_i[r]   = mnew[r];
    }
    // exp, and scatter P into per-wave LDS tile (C/D -> row-major 16x128)
    #pragma unroll
    for (int nt = 0; nt < 8; nt++) {
      #pragma unroll
      for (int r = 0; r < 8; r++) {
        const float p = exp2f((sc[nt][r] - mnew[r]) * 1.44269504f);
        sc[nt][r] = p;
        pw[(lhalf * 8 + r) * PST + nt * 16 + lmod] = (__bf16)p;
      }
    }
    #pragma unroll
    for (int r = 0; r < 8; r++) {
      float s = 0.f;
      #pragma unroll
      for (int nt = 0; nt < 8; nt++) s += sc[nt][r];
      s += __shfl_xor(s, 1);
      s += __shfl_xor(s, 2);
      s += __shfl_xor(s, 4);
      s += __shfl_xor(s, 8);
      l_i[r] = l_i[r] * alpha[r] + s;
    }
    #pragma unroll
    for (int nv = 0; nv < 4; nv++)
      #pragma unroll
      for (int r = 0; r < 8; r++) oacc[nv][r] *= alpha[r];

    // ---- P @ V: P A-frags from LDS, V^T B-frags direct from global/L2 ----
    #pragma unroll
    for (int ks2 = 0; ks2 < 4; ks2++) {
      const __bf16* pp = &pw[lmod * PST + ks2 * 32 + lhalf * 8];
      v16bf pf = cat16(*(const v8bf*)pp, *(const v8bf*)(pp + 16));
      #pragma unroll
      for (int nv = 0; nv < 4; nv++) {
        const __bf16* vb = vp + ((size_t)(nv * 16 + lmod)) * SEQ + kt + ks2 * 32 + lhalf * 8;
        v16bf vf = cat16(*(const v8bf*)vb, *(const v8bf*)(vb + 16));
        oacc[nv] = wmma_bf16(pf, vf, oacc[nv]);
      }
    }
  }

  // ---- normalize + write attn output as bf16 [B, S, DMODEL] ----
  const int b = bh >> 4, h = bh & 15;
  #pragma unroll
  for (int nv = 0; nv < 4; nv++) {
    #pragma unroll
    for (int r = 0; r < 8; r++) {
      const int row = qbase + lhalf * 8 + r;
      const int col = h * DK + nv * 16 + lmod;
      o[((size_t)b * SEQ + row) * DMODEL + col] = (__bf16)(oacc[nv][r] / l_i[r]);
    }
  }
}

// ---------------------------------------------------------------------------
extern "C" void kernel_launch(void* const* d_in, const int* in_sizes, int n_in,
                              void* d_out, int out_size, void* d_ws, size_t ws_size,
                              hipStream_t stream) {
  (void)in_sizes; (void)n_in; (void)out_size; (void)ws_size;
  const float* query = (const float*)d_in[0];
  const float* keyt  = (const float*)d_in[1];
  const float* value = (const float*)d_in[2];
  const float* Wq    = (const float*)d_in[3];
  const float* bq    = (const float*)d_in[4];
  const float* Wk    = (const float*)d_in[5];
  const float* bk    = (const float*)d_in[6];
  const float* Wv    = (const float*)d_in[7];
  const float* bv    = (const float*)d_in[8];
  const float* Wo    = (const float*)d_in[9];
  const float* bo    = (const float*)d_in[10];

  const size_t NACT = (size_t)NBATCH * SEQ * DMODEL;   // 8M elements
  const size_t NWGT = (size_t)DMODEL * DMODEL;         // 1M elements
  char* ws = (char*)d_ws;
  size_t off = 0;
  __bf16* xq = (__bf16*)(ws + off); off += NACT * 2;   // converted activations
  __bf16* xk = (__bf16*)(ws + off); off += NACT * 2;
  __bf16* xv = (__bf16*)(ws + off); off += NACT * 2;
  __bf16* wq = (__bf16*)(ws + off); off += NWGT * 2;   // converted weights
  __bf16* wk = (__bf16*)(ws + off); off += NWGT * 2;
  __bf16* wv = (__bf16*)(ws + off); off += NWGT * 2;
  __bf16* wo = (__bf16*)(ws + off); off += NWGT * 2;
  __bf16* qb = (__bf16*)(ws + off); off += NACT * 2;   // projected Q/K/V^T
  __bf16* kb = (__bf16*)(ws + off); off += NACT * 2;
  __bf16* vb = (__bf16*)(ws + off); off += NACT * 2;
  __bf16* ab = (__bf16*)(ws + off); off += NACT * 2;   // attention output

  const dim3 blk(256);
  const dim3 cga((int)(NACT / 8 / 256));               // 4096 blocks
  const dim3 cgw((int)(NWGT / 8 / 256));               // 512 blocks
  cvt_f32_bf16<<<cga, blk, 0, stream>>>(query, xq, (int)NACT);
  cvt_f32_bf16<<<cga, blk, 0, stream>>>(keyt,  xk, (int)NACT);
  cvt_f32_bf16<<<cga, blk, 0, stream>>>(value, xv, (int)NACT);
  cvt_f32_bf16<<<cgw, blk, 0, stream>>>(Wq, wq, (int)NWGT);
  cvt_f32_bf16<<<cgw, blk, 0, stream>>>(Wk, wk, (int)NWGT);
  cvt_f32_bf16<<<cgw, blk, 0, stream>>>(Wv, wv, (int)NWGT);
  cvt_f32_bf16<<<cgw, blk, 0, stream>>>(Wo, wo, (int)NWGT);

  const dim3 pg(NBATCH * SEQ / 128, DMODEL / 128);     // 64 x 8
  proj_gemm<1><<<pg, blk, 0, stream>>>(xq, wq, bq, (void*)qb);
  proj_gemm<1><<<pg, blk, 0, stream>>>(xk, wk, bk, (void*)kb);
  proj_gemm<2><<<pg, blk, 0, stream>>>(xv, wv, bv, (void*)vb);

  attn_kernel<<<dim3(SEQ / 128, NBATCH * NHEAD), blk, 0, stream>>>(qb, kb, vb, ab);

  proj_gemm<0><<<pg, blk, 0, stream>>>(ab, wo, bo, d_out);
}
